// SelfAttention_35124242546854
// MI455X (gfx1250) — compile-verified, matrix operand order checked by reading, not run on hardware
//
#include <hip/hip_runtime.h>
#include <hip/hip_bf16.h>
#include <math.h>

// Problem constants (B,S,D,H from reference)
#define BB 2
#define SS 2048
#define DD 1024
#define HH 16
#define DHD 64
#define SCALE 0.125f  // DH^-0.5 = 64^-0.5

typedef __attribute__((ext_vector_type(16))) __bf16 v16bf;
typedef __attribute__((ext_vector_type(8)))  float  v8f;

typedef unsigned int       u32;
typedef unsigned long long u64;
typedef u32 u32x4 __attribute__((ext_vector_type(4)));
typedef int i32x4 __attribute__((ext_vector_type(4)));
typedef int i32x8 __attribute__((ext_vector_type(8)));

#if __has_builtin(__builtin_amdgcn_tensor_load_to_lds) && \
    __has_builtin(__builtin_amdgcn_s_wait_tensorcnt)
#define HAVE_TDM 1
#else
#define HAVE_TDM 0
#endif

// ---------------------------------------------------------------------------
// TDM: DMA a 2D f32 tile (tile_h rows x tile_w contiguous elems, row stride
// `stride` elems) from global into LDS at byte offset lds_addr.
// D# packing per CDNA5 ISA 8.3/8.4 (group0 128b, group1 256b, 2D so higher
// groups are zero). This toolchain exposes the 6-arg builtin:
//   (u32x4 g0, i32x8 g1, i32x4 g2, i32x4 g3, i32x8 g4, i32 cpol)
// ---------------------------------------------------------------------------
__device__ __forceinline__ void tdm_load_tile_f32(u32 lds_addr, const float* gptr,
                                                  u32 tile_w, u32 tile_h, u64 stride) {
#if HAVE_TDM
    const u64 ga = (u64)(uintptr_t)gptr;
    u32x4 g0;
    g0.x = 1u;                                              // count=1, user mode
    g0.y = lds_addr;                                        // lds_addr
    g0.z = (u32)(ga & 0xffffffffu);                         // global_addr[31:0]
    g0.w = (u32)((ga >> 32) & 0x01ffffffu) | (2u << 30);    // addr[56:32] | type=2
    i32x8 g1;
    g1[0] = (int)(2u << 16);                                // data_size=4B, mask=0
    g1[1] = (int)((tile_w & 0xffffu) << 16);                // tensor_dim0[15:0]
    g1[2] = (int)((tile_w >> 16) | ((tile_h & 0xffffu) << 16)); // dim0 hi | dim1 lo
    g1[3] = (int)((tile_h >> 16) | ((tile_w & 0xffffu) << 16)); // dim1 hi | tile_dim0
    g1[4] = (int)(tile_h & 0xffffu);                        // tile_dim1, tile_dim2=0
    g1[5] = (int)(u32)(stride & 0xffffffffu);               // dim0_stride[31:0]
    g1[6] = (int)(u32)((stride >> 32) & 0xffffu);           // dim0_stride[47:32]
    g1[7] = 0;
    const i32x4 z4 = {0, 0, 0, 0};
    const i32x8 z8 = {0, 0, 0, 0, 0, 0, 0, 0};
    __builtin_amdgcn_tensor_load_to_lds(g0, g1, z4, z4, z8, 0);
#else
    (void)lds_addr; (void)gptr; (void)tile_w; (void)tile_h; (void)stride;
#endif
}

// Stage helpers: TDM path uses the descriptor above; fallback does a plain
// wave-cooperative copy into the LDS buffer.
__device__ __forceinline__ void stage_tile(float* dstLds, const float* src,
                                           int w, int h, int stride) {
#if HAVE_TDM
    tdm_load_tile_f32((u32)(uintptr_t)dstLds, src, (u32)w, (u32)h, (u64)stride);
#else
    const int lane = threadIdx.x & 31;
    for (int t = lane; t < w * h; t += 32)
        dstLds[t] = src[(size_t)(t / w) * stride + (t % w)];
#endif
}

__device__ __forceinline__ void stage_wait() {
#if HAVE_TDM
    __builtin_amdgcn_s_wait_tensorcnt(0);
#endif
}

// ---------------------------------------------------------------------------
// WMMA fragment helpers (CDNA5 ISA 7.12.2 bf16 layouts, wave32)
//   lanes 0-15:  v0..v3 -> K=0..7,   v4..v7 -> K=16..23
//   lanes 16-31: v0..v3 -> K=8..15,  v4..v7 -> K=24..31
// so per lane the fragment is two contiguous 8-float runs: [8h,8h+8) and
// [16+8h, 24+8h)  ->  4 x global_load_b128.
// ---------------------------------------------------------------------------
__device__ __forceinline__ int kmap(int i, int half) {
    return ((i < 4) ? (2 * i) : (8 + 2 * i)) + 8 * half;
}

__device__ __forceinline__ void cvt8(v16bf& d, int base, float4 c0, float4 c1) {
    d[base + 0] = (__bf16)c0.x; d[base + 1] = (__bf16)c0.y;
    d[base + 2] = (__bf16)c0.z; d[base + 3] = (__bf16)c0.w;
    d[base + 4] = (__bf16)c1.x; d[base + 5] = (__bf16)c1.y;
    d[base + 6] = (__bf16)c1.z; d[base + 7] = (__bf16)c1.w;
}

// A fragment 16x32: tile = &A[tile_row0][k], row stride lda.
__device__ __forceinline__ v16bf load_a_vec(const float* __restrict__ tile, int lda) {
    const int lane = threadIdx.x & 31;
    const int half = lane >> 4;
    const float* rp = tile + (size_t)(lane & 15) * lda;
    const float4* p0 = (const float4*)(rp + 8 * half);
    const float4* p1 = (const float4*)(rp + 16 + 8 * half);
    v16bf a;
    cvt8(a, 0, p0[0], p0[1]);
    cvt8(a, 8, p1[0], p1[1]);
    return a;
}

// B fragment 32x16 where the 16 columns are K-contiguous in memory
// (q.k^T: key vectors), col stride ldn.
__device__ __forceinline__ v16bf load_b_colvec(const float* __restrict__ tile, int ldn) {
    const int lane = threadIdx.x & 31;
    const int half = lane >> 4;
    const float* cp = tile + (size_t)(lane & 15) * ldn;
    const float4* p0 = (const float4*)(cp + 8 * half);
    const float4* p1 = (const float4*)(cp + 16 + 8 * half);
    v16bf b;
    cvt8(b, 0, p0[0], p0[1]);
    cvt8(b, 8, p1[0], p1[1]);
    return b;
}

// B fragment 32x16 from an LDS-resident row-major [32 x 16] f32 tile.
__device__ __forceinline__ v16bf load_b_lds(const float* tile) {
    const int lane = threadIdx.x & 31;
    const int half = lane >> 4;
    const int col  = lane & 15;
    v16bf b;
#pragma unroll
    for (int i = 0; i < 8; ++i) {
        const int kk = kmap(i, half);
        b[2 * i]     = (__bf16)tile[kk * 16 + col];
        b[2 * i + 1] = (__bf16)tile[(kk + 1) * 16 + col];
    }
    return b;
}

// A fragment from f32 scores with on-the-fly softmax (vectorized loads).
__device__ __forceinline__ v16bf load_attn_vec(const float* __restrict__ tile, int lda,
                                               float rmax, float rinv) {
    const int lane = threadIdx.x & 31;
    const int half = lane >> 4;
    const float* rp = tile + (size_t)(lane & 15) * lda;
    const float4* p0 = (const float4*)(rp + 8 * half);
    const float4* p1 = (const float4*)(rp + 16 + 8 * half);
    float4 c[4] = {p0[0], p0[1], p1[0], p1[1]};
    const float* f = (const float*)c;
    v16bf a;
#pragma unroll
    for (int i = 0; i < 16; ++i) a[i] = (__bf16)(__expf(f[i] - rmax) * rinv);
    return a;
}

// ---------------------------------------------------------------------------
// Kernel 1/5: C[M,N] = A[M,K] @ B[K,N] + bias[N]
// Block: one N tile (blockIdx.x), 8 waves = 8 M tiles (blockIdx.y*8+wave).
// B tile (strided operand) is block-uniform -> TDM double-buffered into LDS.
// A fragments: vectorized global b128 loads.
// ---------------------------------------------------------------------------
__global__ void gemm_bias_bf16wmma(const float* __restrict__ A,
                                   const float* __restrict__ Bw,
                                   const float* __restrict__ bias,
                                   float* __restrict__ C,
                                   int M, int N, int K) {
    const int wave  = threadIdx.x >> 5;
    const int lane  = threadIdx.x & 31;
    const int ntile = blockIdx.x;                 // block-uniform
    const int mtile = blockIdx.y * 8 + wave;      // per-wave

    __shared__ float btile[2][32 * 16];           // double-buffered [k][n], 4 KB

    const float* Bt0  = Bw + (size_t)ntile * 16;
    const float* Arow = A + (size_t)mtile * 16 * K;

    if (wave == 0) stage_tile(&btile[0][0], Bt0, 16, 32, N);

    v8f acc = {};
    int idx = 0;
    for (int k = 0; k < K; k += 32, idx ^= 1) {
        if (wave == 0) stage_wait();
        __syncthreads();
        if (wave == 0 && k + 32 < K)
            stage_tile(&btile[idx ^ 1][0], Bt0 + (size_t)(k + 32) * N, 16, 32, N);
        if (k + 32 < K) __builtin_prefetch(Arow + (size_t)lane * K + k + 32, 0, 1);

        v16bf a = load_a_vec(Arow + k, K);
        v16bf b = load_b_lds(&btile[idx][0]);
        acc = __builtin_amdgcn_wmma_f32_16x16x32_bf16(false, a, false, b,
                                                      (short)0, acc, false, false);
        __syncthreads();
    }

    const int half = lane >> 4;
    const int col  = lane & 15;
    const float bv = bias ? bias[ntile * 16 + col] : 0.0f;
#pragma unroll
    for (int r = 0; r < 8; ++r) {
        const int m = mtile * 16 + 8 * half + r;
        C[(size_t)m * N + (size_t)ntile * 16 + col] = acc[r] + bv;
    }
}

// ---------------------------------------------------------------------------
// Kernel 2: scores[b,h,q,k] = (q . k) * SCALE + prev   (written to d_out)
// Both operands are K-contiguous per lane -> pure b128 vector loads.
// qkv layout: [B*S, 3D]; q at +0, k at +D, v at +2D.
// ---------------------------------------------------------------------------
__global__ void scores_kernel(const float* __restrict__ qkv,
                              const float* __restrict__ prev,
                              float* __restrict__ scores) {
    const int bh = blockIdx.z;
    const int b  = bh / HH;
    const int h  = bh % HH;
    const int wave  = threadIdx.x >> 5;
    const int lane  = threadIdx.x & 31;
    const int ntile = blockIdx.x * 8 + wave;  // key tile   (0..127)
    const int mtile = blockIdx.y;             // query tile (0..127)

    const float* qbase = qkv + (size_t)(b * SS + mtile * 16) * (3 * DD) + h * DHD;
    const float* kbase = qkv + (size_t)(b * SS + ntile * 16) * (3 * DD) + DD + h * DHD;

    v8f acc = {};
#pragma unroll
    for (int k = 0; k < DHD; k += 32) {
        v16bf a  = load_a_vec(qbase + k, 3 * DD);
        v16bf bf = load_b_colvec(kbase + k, 3 * DD);
        acc = __builtin_amdgcn_wmma_f32_16x16x32_bf16(false, a, false, bf,
                                                      (short)0, acc, false, false);
    }

    const int half = lane >> 4;
    const int col  = lane & 15;
    const size_t base = ((size_t)bh * SS + mtile * 16) * SS + (size_t)ntile * 16;
#pragma unroll
    for (int r = 0; r < 8; ++r) {
        const size_t idx = base + (size_t)(8 * half + r) * SS + col;
        scores[idx] = acc[r] * SCALE + prev[idx];
    }
}

// ---------------------------------------------------------------------------
// Kernel 3: per-row softmax stats: stats[row] = {max, sum(exp(x-max))}
// ---------------------------------------------------------------------------
__global__ void rowstats_kernel(const float* __restrict__ scores,
                                float* __restrict__ stats) {
    const size_t row = blockIdx.x;
    const float* p = scores + row * (size_t)SS;
    __shared__ float red[256];

    float m = -INFINITY;
    for (int i = threadIdx.x; i < SS; i += 256) m = fmaxf(m, p[i]);
    red[threadIdx.x] = m;
    __syncthreads();
    for (int s = 128; s > 0; s >>= 1) {
        if ((int)threadIdx.x < s)
            red[threadIdx.x] = fmaxf(red[threadIdx.x], red[threadIdx.x + s]);
        __syncthreads();
    }
    const float rmax = red[0];
    __syncthreads();

    float sum = 0.0f;
    for (int i = threadIdx.x; i < SS; i += 256) sum += __expf(p[i] - rmax);
    red[threadIdx.x] = sum;
    __syncthreads();
    for (int s = 128; s > 0; s >>= 1) {
        if ((int)threadIdx.x < s) red[threadIdx.x] += red[threadIdx.x + s];
        __syncthreads();
    }
    if (threadIdx.x == 0) {
        stats[row * 2 + 0] = rmax;
        stats[row * 2 + 1] = red[0];
    }
}

// ---------------------------------------------------------------------------
// Kernel 4: ctx[b,q,h,:] = softmax(scores[b,h,q,:]) @ V[b,:,h,:]
// Block: one dh tile (blockIdx.x) + head/batch (blockIdx.z) -> V tile is
// block-uniform -> TDM double-buffered. Softmax applied on the fly while
// converting the A fragment to bf16 (b128 loads of scores rows).
// ---------------------------------------------------------------------------
__global__ void context_kernel(const float* __restrict__ scores,
                               const float* __restrict__ stats,
                               const float* __restrict__ qkv,
                               float* __restrict__ ctx) {
    const int bh = blockIdx.z;
    const int b  = bh / HH;
    const int h  = bh % HH;
    const int wave  = threadIdx.x >> 5;
    const int lane  = threadIdx.x & 31;
    const int mtile = blockIdx.y * 8 + wave;  // query tile (0..127), per-wave
    const int ntile = blockIdx.x;             // dh tile    (0..3), block-uniform

    __shared__ float vtile[2][32 * 16];       // double-buffered [k][dh], 4 KB

    const int half = lane >> 4;
    const int row  = lane & 15;
    const size_t srow0 = (size_t)bh * SS + (size_t)mtile * 16;
    const float* Sbase = scores + srow0 * SS;
    const float  rmax  = stats[(srow0 + row) * 2 + 0];
    const float  rinv  = 1.0f / stats[(srow0 + row) * 2 + 1];

    const float* Vbase = qkv + (size_t)(b * SS) * (3 * DD) + 2 * DD
                             + h * DHD + ntile * 16;

    if (wave == 0) stage_tile(&vtile[0][0], Vbase, 16, 32, 3 * DD);

    v8f acc = {};
    int idx = 0;
    for (int k = 0; k < SS; k += 32, idx ^= 1) {
        if (wave == 0) stage_wait();
        __syncthreads();
        if (wave == 0 && k + 32 < SS)
            stage_tile(&vtile[idx ^ 1][0], Vbase + (size_t)(k + 32) * (3 * DD),
                       16, 32, 3 * DD);
        if (k + 32 < SS) __builtin_prefetch(Sbase + (size_t)row * SS + k + 32, 0, 1);

        v16bf a  = load_attn_vec(Sbase + k, SS, rmax, rinv);
        v16bf bf = load_b_lds(&vtile[idx][0]);
        acc = __builtin_amdgcn_wmma_f32_16x16x32_bf16(false, a, false, bf,
                                                      (short)0, acc, false, false);
        __syncthreads();
    }

    const int col = lane & 15;
#pragma unroll
    for (int r = 0; r < 8; ++r) {
        const int m = mtile * 16 + 8 * half + r;
        ctx[(size_t)(b * SS + m) * DD + h * DHD + ntile * 16 + col] = acc[r];
    }
}

// ---------------------------------------------------------------------------
// Launcher
// d_in: Q[B,S,D], prev[B,H,S,S], W_qkv[D,3D], b_qkv[3D], W_out[D,D], b_out[D]
// d_out: output[B,S,D] then scores[B,H,S,S]  (reference returns (output, scores))
// ---------------------------------------------------------------------------
extern "C" void kernel_launch(void* const* d_in, const int* in_sizes, int n_in,
                              void* d_out, int out_size, void* d_ws, size_t ws_size,
                              hipStream_t stream) {
    (void)in_sizes; (void)n_in; (void)out_size; (void)ws_size;
    const float* Q    = (const float*)d_in[0];
    const float* prev = (const float*)d_in[1];
    const float* Wqkv = (const float*)d_in[2];
    const float* bqkv = (const float*)d_in[3];
    const float* Wout = (const float*)d_in[4];
    const float* bout = (const float*)d_in[5];

    float* out        = (float*)d_out;
    float* out_scores = out + (size_t)BB * SS * DD;

    float* ws    = (float*)d_ws;
    float* qkv   = ws;                                 // [B*S, 3D]  (50.3 MB)
    float* ctx   = qkv + (size_t)BB * SS * 3 * DD;     // [B*S, D]   (16.8 MB)
    float* stats = ctx + (size_t)BB * SS * DD;         // [B*H*S, 2] (0.5 MB)

    const dim3 blk(256);

    // 1) qkv = Q @ W_qkv + b_qkv     (M=4096, N=3072, K=1024)
    gemm_bias_bf16wmma<<<dim3(3 * DD / 16, BB * SS / 128), blk, 0, stream>>>(
        Q, Wqkv, bqkv, qkv, BB * SS, 3 * DD, DD);
    // 2) scores = q.k^T * scale + prev  -> d_out scores region
    scores_kernel<<<dim3(16, 128, BB * HH), blk, 0, stream>>>(qkv, prev, out_scores);
    // 3) per-row softmax stats
    rowstats_kernel<<<dim3(BB * HH * SS), blk, 0, stream>>>(out_scores, stats);
    // 4) ctx = softmax(scores) @ V
    context_kernel<<<dim3(4, 16, BB * HH), blk, 0, stream>>>(out_scores, stats, qkv, ctx);
    // 5) output = ctx @ W_out + b_out   (M=4096, N=1024, K=1024)
    gemm_bias_bf16wmma<<<dim3(DD / 16, BB * SS / 128), blk, 0, stream>>>(
        ctx, Wout, bout, out, BB * SS, DD, DD);
}
